// TransformerHedge_48473000902848
// MI455X (gfx1250) — compile-verified, matrix operand order checked by reading, not run on hardware
//
#include <hip/hip_runtime.h>
#include <math.h>

// Problem constants (match reference)
#define BB   64     // batch
#define NN   48     // sequence length / autoregressive steps
#define IND  16     // input feature dim (plus 1 prev-delta channel)
#define DD   256    // model dim
#define HH   8      // heads
#define HDD  32     // head dim
#define LL   3      // layers
#define FFD  1024   // feedforward dim
#define NWAVES 16   // 512 threads = 16 wave32s (room for regs, no spills)

typedef __attribute__((ext_vector_type(16))) _Float16 v16h;
typedef __attribute__((ext_vector_type(8)))  _Float16 v8h;
typedef __attribute__((ext_vector_type(8)))  float    v8f;

struct StepArgs {
  const float* features;
  const float* w_in;  const float* b_in;
  const float* b_qkv; const float* b_o;
  const float* ln1_g; const float* ln1_b;
  const float* b_ff1; const float* b_ff2;
  const float* ln2_g; const float* ln2_b;
  const float* b_p1;  const float* w_p2; const float* b_p2;
  const _Float16* wq16; const _Float16* wo16;
  const _Float16* wf1;  const _Float16* wf2; const _Float16* wp1;
  const float* pe;
  float* x; float* y; float* prev; float* out;
  _Float16* hbuf; _Float16* kcache; _Float16* vcache;
};

// Load a 16x32 f16 WMMA A/B fragment from a row-major [rows x ld] f16 matrix.
// Per CDNA5 ISA 16-bit A layout: lanes 0-15 hold row r0+lane with halves
// K=k0+0..7 / k0+16..23; lanes 16-31 hold the same rows with K=k0+8..15 /
// k0+24..31. A row of a row-major W[N,K] is exactly a B-matrix column, so one
// loader serves both operands. Two 16B loads per lane -> b128 vector loads.
__device__ __forceinline__ v16h frag_ld(const _Float16* base, int ld, int r0, int k0) {
  int lane = threadIdx.x & 31;
  const _Float16* p = base + (size_t)(r0 + (lane & 15)) * (size_t)ld
                           + (size_t)(k0 + ((lane & 16) ? 8 : 0));
  v8h lo = *(const v8h*)p;
  v8h hi = *(const v8h*)(p + 16);
  return __builtin_shufflevector(lo, hi, 0,1,2,3,4,5,6,7,8,9,10,11,12,13,14,15);
}

// Two adjacent 16x16 f32 output tiles (n0 and n0+16) of C = A[64,K] * W^T
// (W row-major [Nout,K]). One shared A fragment feeds two independent WMMA
// accumulator chains -> 2x ILP, half the A-side LDS traffic.
template<int K>
__device__ __forceinline__ void gemm_tile2(const _Float16* A, int lda, int m0,
                                           const _Float16* B, int ldb, int n0,
                                           v8f& acc0, v8f& acc1) {
#pragma unroll
  for (int k0 = 0; k0 < K; k0 += 32) {
    v16h a  = frag_ld(A, lda, m0, k0);
    v16h b0 = frag_ld(B, ldb, n0, k0);
    v16h b1 = frag_ld(B, ldb, n0 + 16, k0);
    acc0 = __builtin_amdgcn_wmma_f32_16x16x32_f16(false, a, false, b0,
                                                  (short)0, acc0, false, false);
    acc1 = __builtin_amdgcn_wmma_f32_16x16x32_f16(false, a, false, b1,
                                                  (short)0, acc1, false, false);
  }
}

__device__ __forceinline__ float gelu_f(float v) {
  return 0.5f * v * (1.0f + erff(v * 0.70710678118654752f));
}

__device__ __forceinline__ float wave_sum(float v) {
#pragma unroll
  for (int m = 16; m > 0; m >>= 1) v += __shfl_xor(v, m, 32);   // wave32
  return v;
}

// Row-wise LayerNorm over [BB,DD]; writes f32 (residual stream) + f16 (GEMM A).
__device__ __forceinline__ void layer_norm(const float* src, const float* g,
                                           const float* be, float* dstf,
                                           _Float16* dsth) {
  int wave = threadIdx.x >> 5, lane = threadIdx.x & 31;
  for (int r = wave; r < BB; r += NWAVES) {
    float s = 0.f, ss = 0.f;
#pragma unroll
    for (int j = 0; j < DD / 32; ++j) {
      float v = src[r * DD + lane + j * 32];
      s += v; ss += v * v;
    }
    s = wave_sum(s); ss = wave_sum(ss);
    float mean = s * (1.0f / DD);
    float var  = ss * (1.0f / DD) - mean * mean;
    float rstd = rsqrtf(var + 1e-5f);
#pragma unroll
    for (int j = 0; j < DD / 32; ++j) {
      int c = lane + j * 32;
      float v = (src[r * DD + c] - mean) * rstd * g[c] + be[c];
      dstf[r * DD + c] = v;
      dsth[r * DD + c] = (_Float16)v;
    }
  }
}

// One fused autoregressive decode step (KV-cached): embeds token kk for all 64
// batches, runs 3 encoder layers + head, writes delta into prev[:,kk+1] and
// out[:,kk]. Single workgroup (16 wave32s) so all stage syncs are s_barrier.
__global__ __launch_bounds__(512) void step_kernel(StepArgs sa, int kk) {
  __shared__ __align__(16) unsigned char smem[65536];
  _Float16* xh = (_Float16*)smem;            // [BB][DD] f16 GEMM-A staging
  _Float16* qh = (_Float16*)(smem + 32768);  // [BB][DD] f16 query buffer
  float*    h1 = (float*)(smem + 32768);     // [BB][DD/2] f32 head hidden (aliases qh)

  const int tid = threadIdx.x, wave = tid >> 5, lane = tid & 31;

  // ---- Stage 0: input projection + positional encoding (token kk only) ----
  for (int e = tid; e < BB * DD; e += 512) {
    int b = e >> 8, d = e & (DD - 1);
    const float* wr = sa.w_in + d * (IND + 1);
    float v = sa.b_in[d] + sa.pe[kk * DD + d];
#pragma unroll
    for (int i = 0; i < IND; ++i)
      v += sa.features[((size_t)b * NN + kk) * IND + i] * wr[i];
    v += sa.prev[b * (NN + 1) + kk] * wr[IND];
    sa.x[e] = v;
    xh[e] = (_Float16)v;
  }
  __threadfence_block();
  __syncthreads();

  for (int l = 0; l < LL; ++l) {
    // ---- QKV projection: [64,256] x [768,256]^T; q->LDS, k/v->f16 caches ----
    const _Float16* wq = sa.wq16 + (size_t)l * 3 * DD * DD;
    for (int t = wave; t < 4 * (3 * DD / 32); t += NWAVES) {   // 96 tile-pairs
      int m0 = (t & 3) * 16, n0 = (t >> 2) * 32;
      v8f acc0 = {}, acc1 = {};
      gemm_tile2<DD>(xh, DD, m0, wq, DD, n0, acc0, acc1);
      int mo = (lane & 16) ? 8 : 0;
#pragma unroll
      for (int half = 0; half < 2; ++half) {
        const v8f& acc = half ? acc1 : acc0;
        int nb = n0 + half * 16 + (lane & 15);
        float bias = sa.b_qkv[l * 3 * DD + nb];
#pragma unroll
        for (int i = 0; i < 8; ++i) {
          int m = m0 + mo + i;
          float v = acc[i] + bias;
          if (nb < DD)
            qh[m * DD + nb] = (_Float16)v;
          else if (nb < 2 * DD)
            sa.kcache[(((size_t)l * BB + m) * NN + kk) * DD + (nb - DD)] = (_Float16)v;
          else
            sa.vcache[(((size_t)l * BB + m) * NN + kk) * DD + (nb - 2 * DD)] = (_Float16)v;
        }
      }
    }
    __threadfence_block();
    __syncthreads();

    // ---- Causal attention over KV cache, online softmax, one (b,h)/thread ----
    {
      int b = tid >> 3, h = tid & 7;
      float qv[HDD], cx[HDD];
#pragma unroll
      for (int d = 0; d < HDD; ++d) {
        qv[d] = (float)qh[b * DD + h * HDD + d];
        cx[d] = 0.f;
      }
      float mx = -3.0e38f, sum = 0.f;
      const float scale = 0.17677669529663687f;  // 1/sqrt(HDD)
      for (int t = 0; t <= kk; ++t) {
        const _Float16* kp = sa.kcache + (((size_t)l * BB + b) * NN + t) * DD + h * HDD;
        const _Float16* vp = sa.vcache + (((size_t)l * BB + b) * NN + t) * DD + h * HDD;
        float dot = 0.f;
#pragma unroll
        for (int d = 0; d < HDD; ++d) dot += qv[d] * (float)kp[d];
        dot *= scale;
        float mn = fmaxf(mx, dot);
        float corr = __expf(mx - mn);
        float w = __expf(dot - mn);
        sum = sum * corr + w;
#pragma unroll
        for (int d = 0; d < HDD; ++d) cx[d] = cx[d] * corr + w * (float)vp[d];
        mx = mn;
      }
      float inv = 1.0f / sum;
      __threadfence_block();
      __syncthreads();           // all waves done reading qh/xh from QKV stage
#pragma unroll
      for (int d = 0; d < HDD; ++d)
        xh[b * DD + h * HDD + d] = (_Float16)(cx[d] * inv);  // ctx -> GEMM A
    }
    __threadfence_block();
    __syncthreads();

    // ---- Output projection + residual ----
    const _Float16* wo = sa.wo16 + (size_t)l * DD * DD;
    for (int t = wave; t < 4 * (DD / 32); t += NWAVES) {       // 32 tile-pairs
      int m0 = (t & 3) * 16, n0 = (t >> 2) * 32;
      v8f acc0 = {}, acc1 = {};
      gemm_tile2<DD>(xh, DD, m0, wo, DD, n0, acc0, acc1);
      int mo = (lane & 16) ? 8 : 0;
#pragma unroll
      for (int half = 0; half < 2; ++half) {
        const v8f& acc = half ? acc1 : acc0;
        int nb = n0 + half * 16 + (lane & 15);
        float bias = sa.b_o[l * DD + nb];
#pragma unroll
        for (int i = 0; i < 8; ++i) {
          int m = m0 + mo + i;
          sa.y[m * DD + nb] = sa.x[m * DD + nb] + acc[i] + bias;
        }
      }
    }
    __threadfence_block();
    __syncthreads();

    layer_norm(sa.y, sa.ln1_g + l * DD, sa.ln1_b + l * DD, sa.x, xh);
    __threadfence_block();
    __syncthreads();

    // ---- FF1 + exact GELU -> f16 hidden buffer ----
    const _Float16* w1 = sa.wf1 + (size_t)l * FFD * DD;
    for (int t = wave; t < 4 * (FFD / 32); t += NWAVES) {      // 128 tile-pairs
      int m0 = (t & 3) * 16, n0 = (t >> 2) * 32;
      v8f acc0 = {}, acc1 = {};
      gemm_tile2<DD>(xh, DD, m0, w1, DD, n0, acc0, acc1);
      int mo = (lane & 16) ? 8 : 0;
#pragma unroll
      for (int half = 0; half < 2; ++half) {
        const v8f& acc = half ? acc1 : acc0;
        int nb = n0 + half * 16 + (lane & 15);
        float bias = sa.b_ff1[l * FFD + nb];
#pragma unroll
        for (int i = 0; i < 8; ++i) {
          int m = m0 + mo + i;
          sa.hbuf[(size_t)m * FFD + nb] = (_Float16)gelu_f(acc[i] + bias);
        }
      }
    }
    __threadfence_block();
    __syncthreads();

    // ---- FF2 + residual (K = 1024) ----
    const _Float16* w2 = sa.wf2 + (size_t)l * DD * FFD;
    for (int t = wave; t < 4 * (DD / 32); t += NWAVES) {       // 32 tile-pairs
      int m0 = (t & 3) * 16, n0 = (t >> 2) * 32;
      v8f acc0 = {}, acc1 = {};
      gemm_tile2<FFD>(sa.hbuf, FFD, m0, w2, FFD, n0, acc0, acc1);
      int mo = (lane & 16) ? 8 : 0;
#pragma unroll
      for (int half = 0; half < 2; ++half) {
        const v8f& acc = half ? acc1 : acc0;
        int nb = n0 + half * 16 + (lane & 15);
        float bias = sa.b_ff2[l * DD + nb];
#pragma unroll
        for (int i = 0; i < 8; ++i) {
          int m = m0 + mo + i;
          sa.y[m * DD + nb] = sa.x[m * DD + nb] + acc[i] + bias;
        }
      }
    }
    __threadfence_block();
    __syncthreads();

    layer_norm(sa.y, sa.ln2_g + l * DD, sa.ln2_b + l * DD, sa.x, xh);
    __threadfence_block();
    __syncthreads();
  }

  // ---- Head: h1 = gelu(x @ w_p1^T + b_p1) ----
  for (int t = wave; t < 4 * ((DD / 2) / 32); t += NWAVES) {   // 16 tile-pairs
    int m0 = (t & 3) * 16, n0 = (t >> 2) * 32;
    v8f acc0 = {}, acc1 = {};
    gemm_tile2<DD>(xh, DD, m0, sa.wp1, DD, n0, acc0, acc1);
    int mo = (lane & 16) ? 8 : 0;
#pragma unroll
    for (int half = 0; half < 2; ++half) {
      const v8f& acc = half ? acc1 : acc0;
      int nb = n0 + half * 16 + (lane & 15);
      float bias = sa.b_p1[nb];
#pragma unroll
      for (int i = 0; i < 8; ++i) {
        int m = m0 + mo + i;
        h1[m * (DD / 2) + nb] = gelu_f(acc[i] + bias);
      }
    }
  }
  __threadfence_block();
  __syncthreads();

  // ---- delta = h1 @ w_p2^T + b_p2; feed back into prev, emit output ----
  for (int r = wave; r < BB; r += NWAVES) {
    float s = 0.f;
#pragma unroll
    for (int j = 0; j < (DD / 2) / 32; ++j) {
      int c = lane + j * 32;
      s += h1[r * (DD / 2) + c] * sa.w_p2[c];
    }
    s = wave_sum(s);
    if (lane == 0) {
      float d = s + sa.b_p2[0];
      sa.prev[r * (NN + 1) + kk + 1] = d;
      sa.out[(size_t)r * NN + kk] = d;
    }
  }
}

// ---------- setup kernels ----------
__global__ void cvt_f16_kernel(const float* s, _Float16* d, int n) {
  int i = blockIdx.x * blockDim.x + threadIdx.x;
  if (i < n) d[i] = (_Float16)s[i];
}

__global__ void pe_kernel(float* pe) {
  int i = blockIdx.x * blockDim.x + threadIdx.x;
  if (i < NN * DD) {
    int pos = i / DD, c = i % DD;
    float div = __expf(-(float)(c & ~1) * (logf(10000.0f) / (float)DD));
    float a = (float)pos * div;
    pe[i] = (c & 1) ? cosf(a) : sinf(a);
  }
}

__global__ void zero_kernel(float* p, int n) {
  int i = blockIdx.x * blockDim.x + threadIdx.x;
  if (i < n) p[i] = 0.f;
}

extern "C" void kernel_launch(void* const* d_in, const int* in_sizes, int n_in,
                              void* d_out, int out_size, void* d_ws, size_t ws_size,
                              hipStream_t stream) {
  (void)in_sizes; (void)n_in; (void)out_size; (void)ws_size;
  const float* features = (const float*)d_in[0];
  const float* w_in  = (const float*)d_in[1];
  const float* b_in  = (const float*)d_in[2];
  const float* w_qkv = (const float*)d_in[3];
  const float* b_qkv = (const float*)d_in[4];
  const float* w_o   = (const float*)d_in[5];
  const float* b_o   = (const float*)d_in[6];
  const float* ln1_g = (const float*)d_in[7];
  const float* ln1_b = (const float*)d_in[8];
  const float* w_ff1 = (const float*)d_in[9];
  const float* b_ff1 = (const float*)d_in[10];
  const float* w_ff2 = (const float*)d_in[11];
  const float* b_ff2 = (const float*)d_in[12];
  const float* ln2_g = (const float*)d_in[13];
  const float* ln2_b = (const float*)d_in[14];
  const float* w_p1  = (const float*)d_in[15];
  const float* b_p1  = (const float*)d_in[16];
  const float* w_p2  = (const float*)d_in[17];
  const float* b_p2  = (const float*)d_in[18];

  char* w = (char*)d_ws;
  auto carve = [&](size_t bytes) {
    char* p = w; w += (bytes + 255) & ~(size_t)255; return p;
  };
  float*    pe   = (float*)carve((size_t)NN * DD * sizeof(float));
  float*    x    = (float*)carve((size_t)BB * DD * sizeof(float));
  float*    y    = (float*)carve((size_t)BB * DD * sizeof(float));
  float*    prev = (float*)carve((size_t)BB * (NN + 1) * sizeof(float));
  _Float16* hbuf = (_Float16*)carve((size_t)BB * FFD * 2);
  _Float16* kc   = (_Float16*)carve((size_t)LL * BB * NN * DD * 2);
  _Float16* vc   = (_Float16*)carve((size_t)LL * BB * NN * DD * 2);
  _Float16* wq16 = (_Float16*)carve((size_t)LL * 3 * DD * DD * 2);
  _Float16* wo16 = (_Float16*)carve((size_t)LL * DD * DD * 2);
  _Float16* wf1  = (_Float16*)carve((size_t)LL * FFD * DD * 2);
  _Float16* wf2  = (_Float16*)carve((size_t)LL * DD * FFD * 2);
  _Float16* wp1  = (_Float16*)carve((size_t)(DD / 2) * DD * 2);

  auto cvt = [&](const float* s, _Float16* d, int n) {
    cvt_f16_kernel<<<(n + 255) / 256, 256, 0, stream>>>(s, d, n);
  };
  cvt(w_qkv, wq16, LL * 3 * DD * DD);
  cvt(w_o,   wo16, LL * DD * DD);
  cvt(w_ff1, wf1,  LL * FFD * DD);
  cvt(w_ff2, wf2,  LL * DD * FFD);
  cvt(w_p1,  wp1,  (DD / 2) * DD);
  pe_kernel<<<(NN * DD + 255) / 256, 256, 0, stream>>>(pe);
  zero_kernel<<<(BB * (NN + 1) + 255) / 256, 256, 0, stream>>>(prev, BB * (NN + 1));

  StepArgs sa;
  sa.features = features; sa.w_in = w_in; sa.b_in = b_in;
  sa.b_qkv = b_qkv; sa.b_o = b_o;
  sa.ln1_g = ln1_g; sa.ln1_b = ln1_b;
  sa.b_ff1 = b_ff1; sa.b_ff2 = b_ff2;
  sa.ln2_g = ln2_g; sa.ln2_b = ln2_b;
  sa.b_p1 = b_p1; sa.w_p2 = w_p2; sa.b_p2 = b_p2;
  sa.wq16 = wq16; sa.wo16 = wo16; sa.wf1 = wf1; sa.wf2 = wf2; sa.wp1 = wp1;
  sa.pe = pe; sa.x = x; sa.y = y; sa.prev = prev; sa.out = (float*)d_out;
  sa.hbuf = hbuf; sa.kcache = kc; sa.vcache = vc;

  for (int kk = 0; kk < NN; ++kk)
    step_kernel<<<1, 512, 0, stream>>>(sa, kk);
}
